// RNNClassifier_35940286332962
// MI455X (gfx1250) — matrix operand order repaired and checked by reading, not compiled, not run on hardware
//
#include <hip/hip_runtime.h>

typedef __attribute__((ext_vector_type(16))) _Float16 v16h;
typedef __attribute__((ext_vector_type(8)))  _Float16 v8h;
typedef __attribute__((ext_vector_type(8)))  float    v8f;

#define UNITS   512
#define NGATE   2048          // 4*UNITS
#define BATCH   512
#define TSTEPS  1024
#define NCLS    10
#define R_ON    0.05f
#define ALPHA   0.001f
#define KT      16            // 512 / 32 k-tiles
#define HSTRIDE 520           // padded LDS h row stride in halves (1040B: 16B aligned, bank-rotated)
#define CSTRIDE 516           // LDS c row stride in floats
#define PACK_PER (128*16*32*16)  // halves per packed 512x2048 matrix

// ---------------------------------------------------------------------------
// One-time weight pack: f32 [K=512][N=2048] row-major  ->  f16 WMMA B-fragment
// layout [nt][kt][lane][16 halves]: lane l holds column n = nt*16 + (l&15),
// halves j=0..15 map to K = kt*32 + (l>=16?8:0) + (j&7) + (j>=8?16:0)  (32 contiguous bytes/lane).
// ---------------------------------------------------------------------------
__global__ void pack_weights_f16(const float* __restrict__ U0,
                                 const float* __restrict__ W1,
                                 const float* __restrict__ U1,
                                 _Float16* __restrict__ pk) {
    int idx = blockIdx.x * blockDim.x + threadIdx.x;
    if (idx >= 3 * PACK_PER) return;
    int m = idx / PACK_PER;
    int r = idx % PACK_PER;
    int j  = r & 15;
    int l  = (r >> 4) & 31;
    int kt = (r >> 9) & 15;
    int nt = r >> 13;
    int n = nt * 16 + (l & 15);
    int k = kt * 32 + ((l >> 4) << 3) + (j & 7) + ((j >> 3) << 4);
    const float* src = (m == 0) ? U0 : (m == 1) ? W1 : U1;
    pk[idx] = (_Float16)src[(size_t)k * NGATE + n];
}

// A fragment (16x32 f16) for one k-tile from an LDS h tile (row = ln).
__device__ __forceinline__ v16h a_frag(const _Float16* h, int ln, int hi, int kt) {
    const _Float16* p = h + ln * HSTRIDE + kt * 32 + hi * 8;
    v8h lo = *(const v8h*)p;
    v8h hh = *(const v8h*)(p + 16);
    return __builtin_shufflevector(lo, hh, 0,1,2,3,4,5,6,7,8,9,10,11,12,13,14,15);
}

// B fragment for tile nt, k-tile kt from packed weights (32B contiguous per lane).
__device__ __forceinline__ v16h b_frag(const _Float16* pkw, int nt, int kt, int l) {
    return *(const v16h*)(pkw + (((size_t)nt * KT + kt) * 32 + l) * 16);
}

// acc[g] += h(16x512) @ Wpk(:, gate g, unit subtile ntb) — software-pipelined:
// kt+1's A + 4 B fragments are prefetched while kt's 4 WMMAs issue.
__device__ __forceinline__ void gemm16(v8f* acc, const _Float16* hsrc, const _Float16* pkw,
                                       int ntb, int ln, int hi, int l) {
    v16h a  = a_frag(hsrc, ln, hi, 0);
    v16h b0 = b_frag(pkw, 0 * 32 + ntb, 0, l);
    v16h b1 = b_frag(pkw, 1 * 32 + ntb, 0, l);
    v16h b2 = b_frag(pkw, 2 * 32 + ntb, 0, l);
    v16h b3 = b_frag(pkw, 3 * 32 + ntb, 0, l);
#pragma unroll
    for (int kt = 0; kt < KT - 1; ++kt) {
        v16h an = a_frag(hsrc, ln, hi, kt + 1);
        v16h n0 = b_frag(pkw, 0 * 32 + ntb, kt + 1, l);
        v16h n1 = b_frag(pkw, 1 * 32 + ntb, kt + 1, l);
        v16h n2 = b_frag(pkw, 2 * 32 + ntb, kt + 1, l);
        v16h n3 = b_frag(pkw, 3 * 32 + ntb, kt + 1, l);
        acc[0] = __builtin_amdgcn_wmma_f32_16x16x32_f16(false, a, false, b0, (short)0, acc[0], false, false);
        acc[1] = __builtin_amdgcn_wmma_f32_16x16x32_f16(false, a, false, b1, (short)0, acc[1], false, false);
        acc[2] = __builtin_amdgcn_wmma_f32_16x16x32_f16(false, a, false, b2, (short)0, acc[2], false, false);
        acc[3] = __builtin_amdgcn_wmma_f32_16x16x32_f16(false, a, false, b3, (short)0, acc[3], false, false);
        a = an; b0 = n0; b1 = n1; b2 = n2; b3 = n3;
    }
    acc[0] = __builtin_amdgcn_wmma_f32_16x16x32_f16(false, a, false, b0, (short)0, acc[0], false, false);
    acc[1] = __builtin_amdgcn_wmma_f32_16x16x32_f16(false, a, false, b1, (short)0, acc[1], false, false);
    acc[2] = __builtin_amdgcn_wmma_f32_16x16x32_f16(false, a, false, b2, (short)0, acc[2], false, false);
    acc[3] = __builtin_amdgcn_wmma_f32_16x16x32_f16(false, a, false, b3, (short)0, acc[3], false, false);
}

// ---------------------------------------------------------------------------
// Per-batch-tile phased-LSTM scan. 32 blocks x 256 threads (8 waves).
// Wave w owns units [w*64, w*64+64). Cell update fused per 16-unit subtile
// (only 4 gate acc tiles live); h double-buffered in LDS; c state in LDS.
// ---------------------------------------------------------------------------
__global__ __launch_bounds__(256, 1)
void plstm_scan(const float* __restrict__ inputs, const float* __restrict__ times,
                const float* __restrict__ W0,  const float* __restrict__ b0,
                const float* __restrict__ tau0, const float* __restrict__ s0,
                const float* __restrict__ b1,
                const float* __restrict__ tau1, const float* __restrict__ s1,
                const float* __restrict__ Wfc, const float* __restrict__ bfc,
                const _Float16* __restrict__ pkU0, const _Float16* __restrict__ pkW1,
                const _Float16* __restrict__ pkU1,
                float* __restrict__ out) {
    __shared__ _Float16 h0buf[2][16 * HSTRIDE];
    __shared__ _Float16 h1buf[2][16 * HSTRIDE];
    __shared__ float    c0s[16 * CSTRIDE];
    __shared__ float    c1s[16 * CSTRIDE];
    __shared__ float    wfcs[UNITS * NCLS];
    __shared__ float    bfcs[NCLS];
    __shared__ float    logits[16 * 16];

    const int tid = threadIdx.x;
    const int w   = tid >> 5;        // wave id 0..7
    const int l   = tid & 31;        // lane
    const int ln  = l & 15;          // N-column within tile / A row
    const int hi  = l >> 4;          // half-wave
    const int btile = blockIdx.x * 16;

    for (int i = tid; i < 16 * HSTRIDE; i += 256) {
        h0buf[0][i] = (_Float16)0.f; h0buf[1][i] = (_Float16)0.f;
        h1buf[0][i] = (_Float16)0.f; h1buf[1][i] = (_Float16)0.f;
    }
    for (int i = tid; i < 16 * CSTRIDE; i += 256) { c0s[i] = 0.f; c1s[i] = 0.f; }
    for (int i = tid; i < UNITS * NCLS; i += 256) wfcs[i] = Wfc[i];
    if (tid < NCLS) bfcs[tid] = bfc[tid];
    __syncthreads();

    // per-unit-subtile time-gate params (unit = w*64 + ut*16 + ln)
    float tau0v[4], itau0v[4], s0v[4], tau1v[4], itau1v[4], s1v[4];
    for (int ut = 0; ut < 4; ++ut) {
        int u = w * 64 + ut * 16 + ln;
        tau0v[ut] = tau0[u]; itau0v[ut] = 1.f / tau0v[ut]; s0v[ut] = s0[u];
        tau1v[ut] = tau1[u]; itau1v[ut] = 1.f / tau1v[ut]; s1v[ut] = s1[u];
    }

    for (int t = 0; t < TSTEPS; ++t) {
        const int cur = t & 1, nxt = cur ^ 1;
        const _Float16* hc0 = h0buf[cur];
        _Float16*       hn0 = h0buf[nxt];
        const _Float16* hc1 = h1buf[cur];
        _Float16*       hn1 = h1buf[nxt];

        // per-step per-lane inputs for rows m = r + 8*hi
        float tv[8];
        for (int r = 0; r < 8; ++r) tv[r] = times[(size_t)(btile + r + hi * 8) * TSTEPS + t];
        float xv[8][3];
        for (int r = 0; r < 8; ++r) {
            const float* xp = inputs + ((size_t)(btile + r + hi * 8) * TSTEPS + t) * 3;
            xv[r][0] = xp[0]; xv[r][1] = xp[1]; xv[r][2] = xp[2];
        }

        // =============== layer 0: fused matmul + phased cell update ===============
        for (int ut = 0; ut < 4; ++ut) {
            const int ucol = w * 64 + ut * 16 + ln;
            v8f acc[4];
            for (int g = 0; g < 4; ++g) {
                int n = g * UNITS + ucol;
                float bb = b0[n], wa = W0[n], wb = W0[NGATE + n], wc = W0[2 * NGATE + n];
                v8f a;
                for (int r = 0; r < 8; ++r) a[r] = bb + xv[r][0] * wa + xv[r][1] * wb + xv[r][2] * wc;
                acc[g] = a;
            }
            gemm16(acc, hc0, pkU0, w * 4 + ut, ln, hi, l);   // + h0 @ U0

            float tauv = tau0v[ut], itv = itau0v[ut], sv = s0v[ut];
            for (int r = 0; r < 8; ++r) {
                float ii = 1.f / (1.f + __expf(-acc[0][r]));
                float ff = 1.f / (1.f + __expf(-acc[1][r]));
                float gg = 2.f / (1.f + __expf(-2.f * acc[2][r])) - 1.f;
                float oo = 1.f / (1.f + __expf(-acc[3][r]));
                float cold = c0s[(r + hi * 8) * CSTRIDE + ucol];
                float cc = ff * cold + ii * gg;
                float hcand = oo * (2.f / (1.f + __expf(-2.f * cc)) - 1.f);
                float x = tv[r] - sv;
                float phi = (x - floorf(x * itv) * tauv) * itv;
                float k = (phi < 0.5f * R_ON) ? (2.f / R_ON) * phi
                        : (phi < R_ON)        ? 2.f - (2.f / R_ON) * phi
                                              : ALPHA * phi;
                float hold = (float)hc0[(r + hi * 8) * HSTRIDE + ucol];
                c0s[(r + hi * 8) * CSTRIDE + ucol] = k * cc + (1.f - k) * cold;
                hn0[(r + hi * 8) * HSTRIDE + ucol] = (_Float16)(k * hcand + (1.f - k) * hold);
            }
        }
        __syncthreads();   // h0(next) fully written, all h0(cur) reads done

        // =============== layer 1: z1 = b1 + h0'@W1 + h1@U1, fused update ===============
        for (int ut = 0; ut < 4; ++ut) {
            const int ucol = w * 64 + ut * 16 + ln;
            v8f acc[4];
            for (int g = 0; g < 4; ++g) {
                float bb = b1[g * UNITS + ucol];
                v8f a;
                for (int r = 0; r < 8; ++r) a[r] = bb;
                acc[g] = a;
            }
            gemm16(acc, hn0, pkW1, w * 4 + ut, ln, hi, l);   // + h0' @ W1
            gemm16(acc, hc1, pkU1, w * 4 + ut, ln, hi, l);   // + h1  @ U1

            float tauv = tau1v[ut], itv = itau1v[ut], sv = s1v[ut];
            for (int r = 0; r < 8; ++r) {
                float ii = 1.f / (1.f + __expf(-acc[0][r]));
                float ff = 1.f / (1.f + __expf(-acc[1][r]));
                float gg = 2.f / (1.f + __expf(-2.f * acc[2][r])) - 1.f;
                float oo = 1.f / (1.f + __expf(-acc[3][r]));
                float cold = c1s[(r + hi * 8) * CSTRIDE + ucol];
                float cc = ff * cold + ii * gg;
                float hcand = oo * (2.f / (1.f + __expf(-2.f * cc)) - 1.f);
                float x = tv[r] - sv;
                float phi = (x - floorf(x * itv) * tauv) * itv;
                float k = (phi < 0.5f * R_ON) ? (2.f / R_ON) * phi
                        : (phi < R_ON)        ? 2.f - (2.f / R_ON) * phi
                                              : ALPHA * phi;
                float hold = (float)hc1[(r + hi * 8) * HSTRIDE + ucol];
                c1s[(r + hi * 8) * CSTRIDE + ucol] = k * cc + (1.f - k) * cold;
                hn1[(r + hi * 8) * HSTRIDE + ucol] = (_Float16)(k * hcand + (1.f - k) * hold);
            }
        }
        __syncthreads();   // h1(next) fully written

        // =============== FC + softmax on h1(next) ===============
        if (tid < 16 * NCLS) {
            int b = tid / NCLS, cls = tid % NCLS;
            float a = bfcs[cls];
            const _Float16* hp = hn1 + b * HSTRIDE;
            for (int u2 = 0; u2 < UNITS; ++u2) a += (float)hp[u2] * wfcs[u2 * NCLS + cls];
            logits[b * 16 + cls] = a;
        }
        __syncthreads();
        if (tid < 16) {
            int b = tid;
            float mx = -3.0e38f;
            for (int c2 = 0; c2 < NCLS; ++c2) mx = fmaxf(mx, logits[b * 16 + c2]);
            float e[NCLS], sm = 0.f;
            for (int c2 = 0; c2 < NCLS; ++c2) { e[c2] = __expf(logits[b * 16 + c2] - mx); sm += e[c2]; }
            float inv = 1.f / sm;
            float* op = out + ((size_t)(btile + b) * TSTEPS + t) * NCLS;
            for (int c2 = 0; c2 < NCLS; ++c2) op[c2] = e[c2] * inv;
        }
        __syncthreads();   // protect logits before next step's FC writes
    }
}

// ---------------------------------------------------------------------------
extern "C" void kernel_launch(void* const* d_in, const int* in_sizes, int n_in,
                              void* d_out, int out_size, void* d_ws, size_t ws_size,
                              hipStream_t stream) {
    const float* inputs = (const float*)d_in[0];
    const float* times  = (const float*)d_in[1];
    const float* W0     = (const float*)d_in[2];
    const float* U0     = (const float*)d_in[3];
    const float* b0     = (const float*)d_in[4];
    const float* tau0   = (const float*)d_in[5];
    const float* s0     = (const float*)d_in[6];
    const float* W1     = (const float*)d_in[7];
    const float* U1     = (const float*)d_in[8];
    const float* b1     = (const float*)d_in[9];
    const float* tau1   = (const float*)d_in[10];
    const float* s1     = (const float*)d_in[11];
    const float* Wfc    = (const float*)d_in[12];
    const float* bfc    = (const float*)d_in[13];
    float* out = (float*)d_out;

    _Float16* pk   = (_Float16*)d_ws;            // 3 * 2 MB packed f16 weights
    _Float16* pkU0 = pk;
    _Float16* pkW1 = pk + PACK_PER;
    _Float16* pkU1 = pk + 2 * (size_t)PACK_PER;

    int packThreads = 3 * PACK_PER;
    pack_weights_f16<<<(packThreads + 255) / 256, 256, 0, stream>>>(U0, W1, U1, pk);

    plstm_scan<<<BATCH / 16, 256, 0, stream>>>(inputs, times, W0, b0, tau0, s0,
                                               b1, tau1, s1, Wfc, bfc,
                                               pkU0, pkW1, pkU1, out);
}
